// Model_14843406975077
// MI455X (gfx1250) — matrix operand order repaired and checked by reading, not compiled, not run on hardware
//
#include <hip/hip_runtime.h>

#define EPSF 1e-5f

static constexpr int B_ = 16, L_ = 512, N_ = 256, P_ = 96;
static constexpr int D_ = 512, E_ = 2, DI_ = 1024, DS_ = 16, DR_ = 32, K_ = 4;
static constexpr int M_ = B_ * N_;   // 4096 tokens (one per batch*variate)

typedef __attribute__((ext_vector_type(16))) __bf16 bf16x16;
typedef __attribute__((ext_vector_type(8)))  float  f32x8;

union Frag { uint4 q[2]; bf16x16 v; };

__device__ __forceinline__ float siluf(float x) { return x / (1.f + __expf(-x)); }
__device__ __forceinline__ float geluf(float x) {
  float c = 0.7978845608028654f * (x + 0.044715f * x * x * x);
  return 0.5f * x * (1.f + tanhf(c));
}
__device__ __forceinline__ float softplusf(float x) {
  return (x > 20.f) ? x : __logf(1.f + __expf(x));
}

// ---------------- RevIN stats: mean/std over time per (b,n) ----------------
__global__ __launch_bounds__(256) void k_revin_stats(
    const float* __restrict__ x, float* __restrict__ mean, float* __restrict__ stdv)
{
  int b = blockIdx.x, n = threadIdx.x;                 // block = N_ threads
  float s = 0.f, sq = 0.f;
  for (int l = 0; l < L_; ++l) {
    float v = x[((size_t)b * L_ + l) * N_ + n];
    s += v; sq += v * v;
  }
  float m = s * (1.f / L_);
  float var = sq * (1.f / L_) - m * m;
  mean[b * N_ + n] = m;
  stdv[b * N_ + n] = sqrtf(var + EPSF);
}

// ------- inverted embedding A-matrix: xn transposed to [M, L] in bf16 -------
__global__ __launch_bounds__(256) void k_xnT(
    const float* __restrict__ x, const float* __restrict__ mean,
    const float* __restrict__ stdv, const float* __restrict__ rw,
    const float* __restrict__ rb, __bf16* __restrict__ out)
{
  int id = blockIdx.x * 256 + threadIdx.x;             // over M_*L_
  int l = id % L_, m = id / L_;
  int b = m / N_, n = m % N_;
  float v = x[((size_t)b * L_ + l) * N_ + n];
  float xn = (v - mean[m]) / stdv[m] * rw[n] + rb[n];
  out[(size_t)m * L_ + l] = (__bf16)xn;
}

// --------- weight [K,N] fp32 -> [N,K] bf16 (transposed for B-frags) ---------
__global__ __launch_bounds__(256) void k_wT(
    const float* __restrict__ w, __bf16* __restrict__ out, int Kk, int Nn)
{
  int id = blockIdx.x * 256 + threadIdx.x;
  if (id >= Kk * Nn) return;
  int k = id % Kk, n = id / Kk;
  out[(size_t)n * Kk + k] = (__bf16)w[(size_t)k * Nn + n];
}

__global__ __launch_bounds__(256) void k_f32_to_bf16(
    const float* __restrict__ in, __bf16* __restrict__ out, int n)
{
  int id = blockIdx.x * 256 + threadIdx.x;
  if (id < n) out[id] = (__bf16)in[id];
}

// --------------- LayerNorm over D, write bf16 GEMM operand -----------------
__global__ __launch_bounds__(256) void k_ln_bf16(
    const float* __restrict__ x, const float* __restrict__ w,
    const float* __restrict__ b, __bf16* __restrict__ out, int Dd)
{
  __shared__ float sh[16];
  int row = blockIdx.x;
  const float* xr = x + (size_t)row * Dd;
  float s = 0.f, sq = 0.f;
  for (int j = threadIdx.x; j < Dd; j += 256) { float v = xr[j]; s += v; sq += v * v; }
  for (int off = 16; off > 0; off >>= 1) { s += __shfl_down(s, off); sq += __shfl_down(sq, off); }
  int wave = threadIdx.x >> 5;
  if ((threadIdx.x & 31) == 0) { sh[wave] = s; sh[8 + wave] = sq; }
  __syncthreads();
  if (threadIdx.x == 0) {
    float ts = 0.f, tq = 0.f;
    for (int i = 0; i < 8; ++i) { ts += sh[i]; tq += sh[8 + i]; }
    float m = ts / Dd;
    sh[0] = m;
    sh[1] = rsqrtf(tq / Dd - m * m + EPSF);
  }
  __syncthreads();
  float m = sh[0], is = sh[1];
  for (int j = threadIdx.x; j < Dd; j += 256)
    out[(size_t)row * Dd + j] = (__bf16)((xr[j] - m) * is * w[j] + b[j]);
}

// ----------------- bf16 WMMA GEMM: C = act(A @ B + bias) + resid -----------------
// A: [M,K] bf16 row-major.  Bt: [N,K] bf16 (B transposed).  C/resid: fp32 [M,N].
// 8 waves/block in 4x2 layout; each wave owns a 32x64 tile (2x4 WMMA tiles):
// 8 WMMAs per K=32 step against 12 b128 loads -> ~22 FLOP/byte from L2/WGP$.
// Fragment packing follows the CDNA5 16-bit A(16x32)/B(32x16) VGPR layouts:
//   A lane(row=l15, half): K chunks [k0+8h .. +8) and [k0+16+8h .. +8)
//   B lane(col=l15, half): K chunks [k0+8h .. +8) and [k0+16+8h .. +8)
__global__ __launch_bounds__(256) void k_gemm(
    const __bf16* __restrict__ A, const __bf16* __restrict__ Bt,
    const float* __restrict__ bias, const float* __restrict__ resid,
    float* __restrict__ C, int Nn, int Kk, int act)
{
  const int lane = threadIdx.x & 31;
  const int wave = threadIdx.x >> 5;
  const int half = lane >> 4, l15 = lane & 15;
  const int mBase = blockIdx.x * 128 + (wave & 3) * 32;
  const int nBase = blockIdx.y * 128 + (wave >> 2) * 64;

  f32x8 acc[2][4] = {};

  for (int k0 = 0; k0 < Kk; k0 += 32) {
    Frag a[2], bfr[4];
#pragma unroll
    for (int t = 0; t < 2; ++t) {
      const __bf16* ap = A + (size_t)(mBase + 16 * t + l15) * Kk + k0 + 8 * half;
      a[t].q[0] = *(const uint4*)(ap);
      a[t].q[1] = *(const uint4*)(ap + 16);
    }
#pragma unroll
    for (int t = 0; t < 4; ++t) {
      int n = nBase + 16 * t + l15;
      if (n >= Nn) n = 0;                               // ragged tile: clamp, discard later
      const __bf16* bp = Bt + (size_t)n * Kk + k0 + 8 * half;
      bfr[t].q[0] = *(const uint4*)(bp);
      bfr[t].q[1] = *(const uint4*)(bp + 16);
    }
    // pull the next tiles' lines into WGP$ while WMMAs run (global_prefetch_b8)
    if (k0 + 64 < Kk) {
      __builtin_prefetch(A + (size_t)(mBase + l15) * Kk + k0 + 64, 0, 3);
      int np = nBase + l15;
      if (np < Nn) __builtin_prefetch(Bt + (size_t)np * Kk + k0 + 64, 0, 3);
    }
#pragma unroll
    for (int tm = 0; tm < 2; ++tm)
#pragma unroll
      for (int tn = 0; tn < 4; ++tn)
        acc[tm][tn] = __builtin_amdgcn_wmma_f32_16x16x32_bf16(
            false, a[tm].v, false, bfr[tn].v, (short)0, acc[tm][tn], false, false);
  }

#pragma unroll
  for (int tm = 0; tm < 2; ++tm) {
#pragma unroll
    for (int tn = 0; tn < 4; ++tn) {
      int n = nBase + 16 * tn + l15;
      if (n >= Nn) continue;
      float bv = bias ? bias[n] : 0.f;
#pragma unroll
      for (int r = 0; r < 8; ++r) {
        int m = mBase + 16 * tm + r + 8 * half;         // C/D layout: M = r + 8*half
        float v = acc[tm][tn][r] + bv;
        if (act == 1)      v = siluf(v);
        else if (act == 2) v = geluf(v);
        else if (act == 3) v = softplusf(v);
        size_t idx = (size_t)m * Nn + n;
        if (resid) v += resid[idx];
        C[idx] = v;
      }
    }
  }
}

// --------- causal depthwise conv (K=4) along token axis + SiLU ----------
__global__ __launch_bounds__(256) void k_conv_silu(
    const float* __restrict__ xz, const float* __restrict__ cw,
    const float* __restrict__ cb, float* __restrict__ uf, __bf16* __restrict__ ub)
{
  int id = blockIdx.x * 256 + threadIdx.x;              // over M_*DI_
  int d = id % DI_, m = id / DI_;
  int b = m / N_, n = m % N_;
  float acc = cb[d];
#pragma unroll
  for (int k = 0; k < K_; ++k) {
    int t = n - (K_ - 1) + k;
    if (t >= 0) acc += cw[d * K_ + k] * xz[(size_t)(b * N_ + t) * (2 * DI_) + d];
  }
  float v = siluf(acc);
  uf[(size_t)m * DI_ + d] = v;
  ub[(size_t)m * DI_ + d] = (__bf16)v;
}

// dt slice of dtBC -> bf16 A matrix for the W_dt GEMM
__global__ __launch_bounds__(256) void k_dt_bf16(
    const float* __restrict__ dtbc, __bf16* __restrict__ out)
{
  int id = blockIdx.x * 256 + threadIdx.x;              // over M_*DR_
  int j = id % DR_, m = id / DR_;
  out[(size_t)m * DR_ + j] = (__bf16)dtbc[(size_t)m * 64 + j];
}

// -------------------- selective scan: thread per (b, d) --------------------
__global__ __launch_bounds__(256) void k_scan(
    const float* __restrict__ u, const float* __restrict__ delta,
    const float* __restrict__ dtbc,     // [M,64]: cols 32..47 = B, 48..63 = C
    const float* __restrict__ A_log, const float* __restrict__ Dsk,
    float* __restrict__ y)
{
  int id = blockIdx.x * 256 + threadIdx.x;              // over B_*DI_
  int d = id % DI_, b = id / DI_;
  float Ae[DS_];
#pragma unroll
  for (int s = 0; s < DS_; ++s) Ae[s] = -__expf(A_log[(size_t)d * DS_ + s]);
  float Dv = Dsk[d];
  float h[DS_];
#pragma unroll
  for (int s = 0; s < DS_; ++s) h[s] = 0.f;
  for (int t = 0; t < N_; ++t) {
    size_t m = (size_t)b * N_ + t;
    float dl = delta[m * DI_ + d];
    float uu = u[m * DI_ + d];
    float du = dl * uu;
    const float* Bc = dtbc + m * 64 + 32;
    const float* Cc = dtbc + m * 64 + 48;
    float yv = 0.f;
#pragma unroll
    for (int s = 0; s < DS_; ++s) {
      h[s] = __expf(dl * Ae[s]) * h[s] + du * Bc[s];
      yv += h[s] * Cc[s];
    }
    y[m * DI_ + d] = yv + uu * Dv;
  }
}

// gate: bf16( y * silu(z) )  (z = second half of xz)
__global__ __launch_bounds__(256) void k_gate(
    const float* __restrict__ y, const float* __restrict__ xz, __bf16* __restrict__ out)
{
  int id = blockIdx.x * 256 + threadIdx.x;              // over M_*DI_
  int d = id % DI_, m = id / DI_;
  float z = xz[(size_t)m * (2 * DI_) + DI_ + d];
  out[id] = (__bf16)(y[id] * siluf(z));
}

// --------- projection output [M,P] -> [B,P,N] with RevIN denorm ----------
__global__ __launch_bounds__(256) void k_out_denorm(
    const float* __restrict__ pre, const float* __restrict__ mean,
    const float* __restrict__ stdv, const float* __restrict__ rw,
    const float* __restrict__ rb, float* __restrict__ out)
{
  int id = blockIdx.x * 256 + threadIdx.x;              // over B_*P_*N_
  int n = id % N_;
  int p = (id / N_) % P_;
  int b = id / (N_ * P_);
  int m = b * N_ + n;
  float v = pre[(size_t)m * P_ + p];
  out[id] = (v - rb[n]) / rw[n] * stdv[m] + mean[m];
}

extern "C" void kernel_launch(void* const* d_in, const int* in_sizes, int n_in,
                              void* d_out, int out_size, void* d_ws, size_t ws_size,
                              hipStream_t stream)
{
  (void)in_sizes; (void)n_in; (void)out_size; (void)ws_size;
  const float* x_enc   = (const float*)d_in[0];
  const float* revin_w = (const float*)d_in[1];
  const float* revin_b = (const float*)d_in[2];
  const float* emb_W   = (const float*)d_in[3];
  const float* emb_b   = (const float*)d_in[4];
  const float* ln1_w   = (const float*)d_in[5];
  const float* ln1_b   = (const float*)d_in[6];
  const float* W_in    = (const float*)d_in[7];
  const float* conv_w  = (const float*)d_in[8];
  const float* conv_b  = (const float*)d_in[9];
  const float* W_x     = (const float*)d_in[10];
  const float* W_dt    = (const float*)d_in[11];
  const float* b_dt    = (const float*)d_in[12];
  const float* A_log   = (const float*)d_in[13];
  const float* D_skip  = (const float*)d_in[14];
  const float* W_out   = (const float*)d_in[15];
  const float* b_out   = (const float*)d_in[16];
  const float* ln2_w   = (const float*)d_in[17];
  const float* ln2_b   = (const float*)d_in[18];
  const float* mlp_W1  = (const float*)d_in[19];
  const float* mlp_b1  = (const float*)d_in[20];
  const float* mlp_W2  = (const float*)d_in[21];
  const float* mlp_b2  = (const float*)d_in[22];
  const float* proj_W  = (const float*)d_in[23];
  const float* proj_b  = (const float*)d_in[24];
  float* out = (float*)d_out;

  // ---- bump-allocate workspace (~107 MB; fits the 192 MB L2) ----
  uintptr_t p = (uintptr_t)d_ws;
  auto take = [&](size_t bytes) -> void* {
    uintptr_t r = (p + 255) & ~(uintptr_t)255;
    p = r + bytes;
    return (void*)r;
  };
  float*  f_mean  = (float*)take((size_t)M_ * 4);
  float*  f_std   = (float*)take((size_t)M_ * 4);
  __bf16* bf_A    = (__bf16*)take((size_t)M_ * 2048 * 2);   // GEMM A staging (bf16)
  __bf16* bf_W    = (__bf16*)take((size_t)2048 * 512 * 2);  // transposed weight staging
  float*  f_tmp   = (float*)take((size_t)M_ * 2048 * 4);    // xz / mlp hidden / proj out
  float*  f_tok   = (float*)take((size_t)M_ * D_ * 4);
  float*  f_u     = (float*)take((size_t)M_ * DI_ * 4);
  float*  f_delta = (float*)take((size_t)M_ * DI_ * 4);
  float*  f_y     = (float*)take((size_t)M_ * DI_ * 4);
  float*  f_dtbc  = (float*)take((size_t)M_ * 64 * 4);

  auto gemm = [&](const __bf16* A, const __bf16* Bt, const float* bias,
                  const float* resid, float* C, int Nn, int Kk, int act) {
    dim3 g(M_ / 128, (Nn + 127) / 128);
    k_gemm<<<g, 256, 0, stream>>>(A, Bt, bias, resid, C, Nn, Kk, act);
  };
  auto wT = [&](const float* w, int Kk, int Nn) {
    int tot = Kk * Nn;
    k_wT<<<(tot + 255) / 256, 256, 0, stream>>>(w, bf_W, Kk, Nn);
  };

  // RevIN stats + inverted series embedding (tok = xn^T @ emb_W + emb_b)
  k_revin_stats<<<B_, N_, 0, stream>>>(x_enc, f_mean, f_std);
  k_xnT<<<(M_ * L_) / 256, 256, 0, stream>>>(x_enc, f_mean, f_std, revin_w, revin_b, bf_A);
  wT(emb_W, L_, D_);
  gemm(bf_A, bf_W, emb_b, nullptr, f_tok, D_, L_, 0);

  for (int i = 0; i < E_; ++i) {
    // -------- Mamba branch --------
    k_ln_bf16<<<M_, 256, 0, stream>>>(f_tok, ln1_w + i * D_, ln1_b + i * D_, bf_A, D_);
    wT(W_in + (size_t)i * D_ * 2 * DI_, D_, 2 * DI_);
    gemm(bf_A, bf_W, nullptr, nullptr, f_tmp, 2 * DI_, D_, 0);             // xz
    k_conv_silu<<<(M_ * DI_) / 256, 256, 0, stream>>>(
        f_tmp, conv_w + (size_t)i * DI_ * K_, conv_b + (size_t)i * DI_, f_u, bf_A);
    wT(W_x + (size_t)i * DI_ * (DR_ + 2 * DS_), DI_, DR_ + 2 * DS_);
    gemm(bf_A, bf_W, nullptr, nullptr, f_dtbc, DR_ + 2 * DS_, DI_, 0);     // dt|B|C
    k_dt_bf16<<<(M_ * DR_) / 256, 256, 0, stream>>>(f_dtbc, bf_A);
    wT(W_dt + (size_t)i * DR_ * DI_, DR_, DI_);
    gemm(bf_A, bf_W, b_dt + (size_t)i * DI_, nullptr, f_delta, DI_, DR_, 3); // softplus
    k_scan<<<(B_ * DI_) / 256, 256, 0, stream>>>(
        f_u, f_delta, f_dtbc, A_log + (size_t)i * DI_ * DS_,
        D_skip + (size_t)i * DI_, f_y);
    k_gate<<<(M_ * DI_) / 256, 256, 0, stream>>>(f_y, f_tmp, bf_A);
    wT(W_out + (size_t)i * DI_ * D_, DI_, D_);
    gemm(bf_A, bf_W, b_out + (size_t)i * D_, f_tok, f_tok, D_, DI_, 0);    // tok += mamba

    // -------- MLP branch --------
    k_ln_bf16<<<M_, 256, 0, stream>>>(f_tok, ln2_w + i * D_, ln2_b + i * D_, bf_A, D_);
    wT(mlp_W1 + (size_t)i * D_ * 4 * D_, D_, 4 * D_);
    gemm(bf_A, bf_W, mlp_b1 + (size_t)i * 4 * D_, nullptr, f_tmp, 4 * D_, D_, 2); // gelu
    k_f32_to_bf16<<<(M_ * 4 * D_) / 256, 256, 0, stream>>>(f_tmp, bf_A, M_ * 4 * D_);
    wT(mlp_W2 + (size_t)i * 4 * D_ * D_, 4 * D_, D_);
    gemm(bf_A, bf_W, mlp_b2 + (size_t)i * D_, f_tok, f_tok, D_, 4 * D_, 0); // tok += mlp
  }

  // projection + RevIN denorm -> [B,P,N]
  k_f32_to_bf16<<<(M_ * D_) / 256, 256, 0, stream>>>(f_tok, bf_A, M_ * D_);
  wT(proj_W, D_, P_);
  gemm(bf_A, bf_W, proj_b, nullptr, f_tmp, P_, D_, 0);
  k_out_denorm<<<(B_ * P_ * N_) / 256, 256, 0, stream>>>(
      f_tmp, f_mean, f_std, revin_w, revin_b, out);
}
